// MultiSupConLoss2_23381801959425
// MI455X (gfx1250) — 16-bit and 8-bit matrix operand order repaired and checked by reading, compiled, not run on hardware
//
#include <hip/hip_runtime.h>
#include <hip/hip_bf16.h>
#include <stdint.h>

#define BB   4096
#define VV   2
#define DD   256
#define CC   80
#define NN   (BB*VV)
#define LPAD 128
#define NWAVES 8
#define INVT (1.0f / 0.07f)

typedef __attribute__((ext_vector_type(16))) _Float16 v16h;
typedef __attribute__((ext_vector_type(8)))  _Float16 v8h;
typedef __attribute__((ext_vector_type(8)))  float    v8f;
typedef __attribute__((ext_vector_type(8)))  int      v8i;
typedef __attribute__((ext_vector_type(4)))  int      v4i;
typedef __attribute__((ext_vector_type(2)))  int      v2i;


__device__ __forceinline__ v16h load_a_f16(const _Float16* rowK, bool hi) {
  const _Float16* p = rowK + (hi ? 16 : 0);
  v8h x0 = *(const v8h*)(p);
  v8h x1 = *(const v8h*)(p + 8);
  return __builtin_shufflevector(x0, x1, 0,1,2,3,4,5,6,7,8,9,10,11,12,13,14,15);
}

__device__ __forceinline__ v8i load_a_u8(const unsigned char* p) {
  v2i x0 = *(const v2i*)(p);
  v2i x1 = *(const v2i*)(p + 16);
  v2i x2 = *(const v2i*)(p + 32);
  v2i x3 = *(const v2i*)(p + 48);
  v8i r;
  r[0]=x0[0]; r[1]=x0[1]; r[2]=x1[0]; r[3]=x1[1];
  r[4]=x2[0]; r[5]=x2[1]; r[6]=x3[0]; r[7]=x3[1];
  return r;
}

__device__ __forceinline__ v8i load_b_u8(const unsigned char* p) {
  v2i x0 = *(const v2i*)(p);
  v2i x1 = *(const v2i*)(p + 16);
  v2i x2 = *(const v2i*)(p + 32);
  v2i x3 = *(const v2i*)(p + 48);
  v8i r;
  r[0]=x0[0]; r[1]=x0[1]; r[2]=x1[0]; r[3]=x1[1];
  r[4]=x2[0]; r[5]=x2[1]; r[6]=x3[0]; r[7]=x3[1];
  return r;
}

__device__ __forceinline__ void load_bf_tile(v16h (&buf)[8], const _Float16* brow) {
#pragma unroll
  for (int ck = 0; ck < 8; ++ck) buf[ck] = *(const v16h*)(brow + ck * 32);
}

__global__ __launch_bounds__(256) void conv_f16_kernel(const float* __restrict__ f,
                                                       _Float16* __restrict__ ct) {
  int idx = blockIdx.x * 256 + threadIdx.x;
  int n = idx >> 8;
  int d = idx & (DD - 1);
  int vv = n >> 12;
  int b  = n & (BB - 1);
  ct[idx] = (_Float16)f[((size_t)b * VV + vv) * DD + d];
}

__global__ __launch_bounds__(32) void pack_labels_kernel(const float* __restrict__ labels,
                                                         unsigned char* __restrict__ lab,
                                                         int* __restrict__ s) {
  int b = blockIdx.x;
  int t = threadIdx.x;
  unsigned int word = 0;
  int cnt = 0;
#pragma unroll
  for (int k = 0; k < 4; ++k) {
    int c = t * 4 + k;
    unsigned int u = 0;
    if (c < CC) u = (labels[(size_t)b * CC + c] > 0.5f) ? 1u : 0u;
    cnt += (int)u;
    word |= u << (8 * k);
  }
  ((unsigned int*)lab)[b * (LPAD / 4) + t] = word;
#pragma unroll
  for (int m = 1; m <= 16; m <<= 1) cnt += __shfl_xor(cnt, m, 32);
  if (t == 0) s[b] = cnt;
}

__global__ __launch_bounds__(32 * NWAVES) void supcon_main_kernel(
    const _Float16* __restrict__ ct, const unsigned char* __restrict__ lab,
    const int* __restrict__ s, float* __restrict__ partials) {
  __shared__ float sP[16][NWAVES];
  __shared__ float sN[16][NWAVES];

  const int  tid  = threadIdx.x;
  const int  w    = tid >> 5;
  const int  lane = tid & 31;
  const int  lr   = lane & 15;
  const bool hi   = lane >= 16;
  const int  m0   = blockIdx.x * 16;
  const int  hofs = hi ? 8 : 0;

  v16h af[8];
  const _Float16* arow = ct + (size_t)(m0 + lr) * DD;
#pragma unroll
  for (int ck = 0; ck < 8; ++ck) af[ck] = load_a_f16(arow + ck * 32, hi);

  v8i al0, al1;
  {
    const unsigned char* alrow = lab + (size_t)(m0 + lr) * LPAD + (hi ? 8 : 0);
    al0 = load_a_u8(alrow);
    al1 = load_a_u8(alrow + 64);
  }

  int si[8];
#pragma unroll
  for (int v = 0; v < 8; ++v) si[v] = s[m0 + v + hofs];

  float accP[8], accN[8];
#pragma unroll
  for (int v = 0; v < 8; ++v) { accP[v] = 0.f; accN[v] = 0.f; }

  const int stride = 16 * NWAVES;

  int b0 = w * 16;
  v8i  bl0, bl1;
  int  sb;
  v16h bfA[8], bfB[8];
  {
    const unsigned char* blrow = lab + (size_t)(b0 + lr) * LPAD + (hi ? 8 : 0);
    bl0 = load_b_u8(blrow);
    bl1 = load_b_u8(blrow + 64);
    sb  = s[b0 + lr];
    load_bf_tile(bfA, ct + (size_t)(b0 + lr) * DD + (hi ? 16 : 0));
  }

  for (; b0 < BB; ) {
    const int b0c = b0;
    const int sbc = sb;
    const int bn  = b0 + stride;

    v8i ic = {};
    ic = __builtin_amdgcn_wmma_i32_16x16x64_iu8(false, al0, false, bl0, ic, false, false);
    ic = __builtin_amdgcn_wmma_i32_16x16x64_iu8(false, al1, false, bl1, ic, false, false);

    unsigned pmask = 0;
#pragma unroll
    for (int v = 0; v < 8; ++v) {
      int in = ic[v];
      if (2 * in > si[v] + sbc - in) pmask |= 1u << v;
    }

    load_bf_tile(bfB, ct + (size_t)(BB + b0c + lr) * DD + (hi ? 16 : 0));

    {
      v8f dc = {};
#pragma unroll
      for (int ck = 0; ck < 8; ++ck)
        dc = __builtin_amdgcn_wmma_f32_16x16x32_f16(false, af[ck], false, bfA[ck],
                                                    (short)0, dc, false, false);
#pragma unroll
      for (int v = 0; v < 8; ++v) {
        if (b0c + lr != m0 + v + hofs) {
          float e = __expf(dc[v] * INVT - INVT);
          if ((pmask >> v) & 1u) accP[v] += e; else accN[v] += e;
        }
      }
    }

    if (bn < BB) {
      const unsigned char* blrow = lab + (size_t)(bn + lr) * LPAD + (hi ? 8 : 0);
      bl0 = load_b_u8(blrow);
      bl1 = load_b_u8(blrow + 64);
      sb  = s[bn + lr];
      load_bf_tile(bfA, ct + (size_t)(bn + lr) * DD + (hi ? 16 : 0));
    }

    {
      v8f dc = {};
#pragma unroll
      for (int ck = 0; ck < 8; ++ck)
        dc = __builtin_amdgcn_wmma_f32_16x16x32_f16(false, af[ck], false, bfB[ck],
                                                    (short)0, dc, false, false);
#pragma unroll
      for (int v = 0; v < 8; ++v) {
        float e = __expf(dc[v] * INVT - INVT);
        if ((pmask >> v) & 1u) accP[v] += e; else accN[v] += e;
      }
    }

    b0 = bn;
  }

#pragma unroll
  for (int v = 0; v < 8; ++v) {
#pragma unroll
    for (int m = 1; m <= 8; m <<= 1) {
      accP[v] += __shfl_xor(accP[v], m, 32);
      accN[v] += __shfl_xor(accN[v], m, 32);
    }
  }
  if (lr == 0) {
#pragma unroll
    for (int v = 0; v < 8; ++v) {
      int r = v + hofs;
      sP[r][w] = accP[v];
      sN[r][w] = accN[v];
    }
  }
  __syncthreads();

  if (tid < 32) {
    float ls = 0.f, cn = 0.f;
    if (tid < 16) {
      float ps = 0.f, ns = 0.f;
#pragma unroll
      for (int k = 0; k < NWAVES; ++k) { ps += sP[tid][k]; ns += sN[tid][k]; }
      if (ps > 0.f) {
        ls = -__logf(ps / (ps + ns + 1e-8f));
        cn = 1.f;
      }
    }
#pragma unroll
    for (int m = 1; m <= 8; m <<= 1) {
      ls += __shfl_xor(ls, m, 32);
      cn += __shfl_xor(cn, m, 32);
    }
    if (tid == 0) {
      partials[blockIdx.x * 2 + 0] = ls;
      partials[blockIdx.x * 2 + 1] = cn;
    }
  }
}

__global__ void finalize_kernel(const float* __restrict__ partials, float* __restrict__ out) {
  float ls = 0.f, cn = 0.f;
  for (int i = 0; i < BB / 16; ++i) {
    ls += partials[2 * i + 0];
    cn += partials[2 * i + 1];
  }
  out[0] = ls / fmaxf(cn, 1.f);
}

extern "C" void kernel_launch(void* const* d_in, const int* in_sizes, int n_in,
                              void* d_out, int out_size, void* d_ws, size_t ws_size,
                              hipStream_t stream) {
  (void)in_sizes; (void)n_in; (void)out_size; (void)ws_size;
  const float* features = (const float*)d_in[0];
  const float* labels   = (const float*)d_in[1];
  float* out = (float*)d_out;

  char* ws = (char*)d_ws;
  _Float16*      ct       = (_Float16*)(ws);
  unsigned char* lab      = (unsigned char*)(ws + (size_t)NN * DD * 2);
  int*           s        = (int*)(ws + (size_t)NN * DD * 2 + (size_t)BB * LPAD);
  float*         partials = (float*)(ws + (size_t)NN * DD * 2 + (size_t)BB * LPAD
                                        + (size_t)BB * sizeof(int));

  conv_f16_kernel   <<<(NN * DD) / 256, 256, 0, stream>>>(features, ct);
  pack_labels_kernel<<<BB, 32, 0, stream>>>(labels, lab, s);
  supcon_main_kernel<<<BB / 16, 32 * NWAVES, 0, stream>>>(ct, lab, s, partials);
  finalize_kernel   <<<1, 1, 0, stream>>>(partials, out);
}
